// AMN_7464653161002
// MI455X (gfx1250) — compile-verified, matrix operand order checked by reading, not run on hardware
//
#include <hip/hip_runtime.h>
#include <math.h>

typedef __attribute__((ext_vector_type(16))) _Float16 v16h;
typedef __attribute__((ext_vector_type(8)))  float    v8f;

// ---------------------------------------------------------------------------
// A/B fragment K mapping for V_WMMA_F32_16X16X32_F16 (wave32):
//   lane group g = lane>>4; half element e (0..15) holds
//   K = kbase + g*8 + e            for e in [0,8)
//   K = kbase + g*8 + e + 8        for e in [8,16)
// A: lane owns row m = lane&15. B (column-major load): lane owns col n = lane&15.
// C/D: element f holds row m = f + 8*g, col n = lane&15.
// ---------------------------------------------------------------------------

__device__ __forceinline__ v16h load_frag_rowK(const float* __restrict__ row,
                                               int kbase, int kmax, int g,
                                               bool valid) {
  v16h f;
#pragma unroll
  for (int e = 0; e < 16; ++e) {
    int k = kbase + g * 8 + e + ((e < 8) ? 0 : 8);
    float v = (valid && k < kmax) ? row[k] : 0.f;
    f[e] = (_Float16)v;
  }
  return f;
}

// ---------------------------------------------------------------------------
// Implicit-GEMM conv3x3 (pad=1, no bias), optional SiLU.
// grid: (Np/32, Co/16, B), block: 32 (one wave).
// Each wave: one weight A-fragment shared by TWO pixel subtiles (2 WMMAs).
// ---------------------------------------------------------------------------
__global__ void conv_wmma_kernel(const float* __restrict__ in,
                                 const float* __restrict__ w,
                                 float* __restrict__ out,
                                 int Ci, int Hi, int Wi,
                                 int Co, int Ho, int Wo,
                                 int stride, int act) {
  const int lane = threadIdx.x;
  const int g = lane >> 4, nn = lane & 15;
  const int b = blockIdx.z;
  const int Np = Ho * Wo;
  const int pix0 = blockIdx.x * 32 + nn;     // B-frag columns, subtile 0
  const int pix1 = pix0 + 16;                // subtile 1
  const int oy0 = pix0 / Wo, ox0 = pix0 % Wo;
  const int oy1 = pix1 / Wo, ox1 = pix1 % Wo;
  const int co = blockIdx.y * 16 + nn;       // output channel (A-frag row)
  const int K = Ci * 9;

  const float* __restrict__ wrow = w + (size_t)co * K;
  const float* __restrict__ inb = in + (size_t)b * Ci * Hi * Wi;

  v8f acc0 = {}, acc1 = {};
  for (int k0 = 0; k0 < K; k0 += 32) {
    v16h a = load_frag_rowK(wrow, k0, K, g, true);
    v16h bi0, bi1;
#pragma unroll
    for (int e = 0; e < 16; ++e) {
      int k = k0 + g * 8 + e + ((e < 8) ? 0 : 8);
      float v0 = 0.f, v1 = 0.f;
      if (k < K) {
        int ci = k / 9, tap = k - ci * 9;
        int dy = tap / 3 - 1, dx = (tap % 3) - 1;
        const float* cin = inb + (size_t)ci * Hi * Wi;
        int iy0 = oy0 * stride + dy, ix0 = ox0 * stride + dx;
        int iy1 = oy1 * stride + dy, ix1 = ox1 * stride + dx;
        if ((unsigned)iy0 < (unsigned)Hi && (unsigned)ix0 < (unsigned)Wi)
          v0 = cin[(size_t)iy0 * Wi + ix0];
        if ((unsigned)iy1 < (unsigned)Hi && (unsigned)ix1 < (unsigned)Wi)
          v1 = cin[(size_t)iy1 * Wi + ix1];
      }
      bi0[e] = (_Float16)v0;
      bi1[e] = (_Float16)v1;
    }
    // reuse_a on the first: A operand cached for the next (identical) WMMA
    acc0 = __builtin_amdgcn_wmma_f32_16x16x32_f16(false, a, false, bi0,
                                                  (short)0, acc0, true, false);
    acc1 = __builtin_amdgcn_wmma_f32_16x16x32_f16(false, a, false, bi1,
                                                  (short)0, acc1, false, false);
  }

  float* __restrict__ outb = out + (size_t)b * Co * Np;
#pragma unroll
  for (int f = 0; f < 8; ++f) {
    int c = blockIdx.y * 16 + f + 8 * g;
    float v0 = acc0[f], v1 = acc1[f];
    if (act) {
      v0 = v0 * (1.f / (1.f + expf(-v0)));   // SiLU
      v1 = v1 * (1.f / (1.f + expf(-v1)));
    }
    outb[(size_t)c * Np + pix0] = v0;
    outb[(size_t)c * Np + pix1] = v1;
  }
}

// ---------------------------------------------------------------------------
// Tiled WMMA GEMM: OUT[m, n] = sum_k A[m,k] * W[n,k] (+bias +resid, act)
// Macro-tile 64(M) x 64(N), K-step 32. Block 256 threads = 8 waves; wave w
// owns subtile (mi = w>>1, ni in {2(w&1), 2(w&1)+1}) -> 2 WMMAs sharing A.
// Operands staged global->LDS with GLOBAL_LOAD_ASYNC_TO_LDS_B128 (ASYNCcnt),
// uniform scalar fallback for the K tail (zero fill).
// Requires M % 64 == 0 (true here: M = 8192). act: 0 none, 1 relu.
// ---------------------------------------------------------------------------
__global__ void gemm_wmma_kernel(const float* __restrict__ A, int lda,
                                 const float* __restrict__ W,
                                 const float* __restrict__ bias,
                                 const float* __restrict__ resid,
                                 float* __restrict__ OUT, int ldo,
                                 int M, int N, int K, int act) {
  __shared__ float At[64][36];   // row stride 144B: 16B-aligned, LDS-bank-clean
  __shared__ float Wt[64][36];

  const int wave = threadIdx.x >> 5;
  const int lane = threadIdx.x & 31;
  const int g = lane >> 4, nn = lane & 15;
  const int mi = wave >> 1;                 // 0..3
  const int nb = (wave & 1) * 2;            // 0 or 2 (two 16-col subtiles)
  const int tm = blockIdx.y, tn = blockIdx.x;

  // staging role: thread covers 4 contiguous floats in rows r0 and r0+32
  const int r0 = threadIdx.x >> 3;          // 0..31
  const int c0 = (threadIdx.x & 7) * 4;     // 0,4,...,28

  v8f acc0 = {}, acc1 = {};

  for (int k0 = 0; k0 < K; k0 += 32) {
    __syncthreads();                        // protect LDS from prior reads
    const bool tail = (k0 + 32 > K);
    if (!tail) {
#pragma unroll
      for (int p = 0; p < 2; ++p) {
        const int r = r0 + p * 32;
        const float* ga = A + (size_t)(tm * 64 + r) * lda + k0 + c0;
        unsigned la = (unsigned)(uintptr_t)&At[r][c0];
        asm volatile("global_load_async_to_lds_b128 %0, %1, off"
                     :: "v"(la), "v"(ga) : "memory");
        const int nrow = tn * 64 + r;
        if (nrow < N) {
          const float* gw = W + (size_t)nrow * K + k0 + c0;
          unsigned lw = (unsigned)(uintptr_t)&Wt[r][c0];
          asm volatile("global_load_async_to_lds_b128 %0, %1, off"
                       :: "v"(lw), "v"(gw) : "memory");
        }
      }
      asm volatile("s_wait_asynccnt 0x0" ::: "memory");
    } else {
      // uniform K-tail: zero-fill via plain DS stores
#pragma unroll
      for (int p = 0; p < 2; ++p) {
        const int r = r0 + p * 32;
        const float* ga = A + (size_t)(tm * 64 + r) * lda;
        const int nrow = tn * 64 + r;
        const float* gw = W + (size_t)nrow * K;
#pragma unroll
        for (int j = 0; j < 4; ++j) {
          const int k = k0 + c0 + j;
          At[r][c0 + j] = (k < K) ? ga[k] : 0.f;
          Wt[r][c0 + j] = (k < K && nrow < N) ? gw[k] : 0.f;
        }
      }
    }
    __syncthreads();

    v16h a, b0, b1;
#pragma unroll
    for (int e = 0; e < 16; ++e) {
      const int k = g * 8 + e + ((e < 8) ? 0 : 8);
      a[e]  = (_Float16)At[mi * 16 + nn][k];
      b0[e] = (_Float16)Wt[nb * 16 + nn][k];
      b1[e] = (_Float16)Wt[nb * 16 + 16 + nn][k];
    }
    // reuse_a on the first: A operand cached for the next (identical) WMMA
    acc0 = __builtin_amdgcn_wmma_f32_16x16x32_f16(false, a, false, b0,
                                                  (short)0, acc0, true, false);
    acc1 = __builtin_amdgcn_wmma_f32_16x16x32_f16(false, a, false, b1,
                                                  (short)0, acc1, false, false);
  }

  const int n0 = tn * 64 + nb * 16 + nn;
  const int n1 = n0 + 16;
  const float bv0 = (bias && n0 < N) ? bias[n0] : 0.f;
  const float bv1 = (bias && n1 < N) ? bias[n1] : 0.f;
#pragma unroll
  for (int f = 0; f < 8; ++f) {
    const int m = tm * 64 + mi * 16 + f + 8 * g;   // always < M (M%64==0)
    if (n0 < N) {
      float v = acc0[f] + bv0;
      if (resid) v += resid[(size_t)m * ldo + n0];
      if (act == 1) v = fmaxf(v, 0.f);
      OUT[(size_t)m * ldo + n0] = v;
    }
    if (n1 < N) {
      float v = acc1[f] + bv1;
      if (resid) v += resid[(size_t)m * ldo + n1];
      if (act == 1) v = fmaxf(v, 0.f);
      OUT[(size_t)m * ldo + n1] = v;
    }
  }
}

// ---------------------------------------------------------------------------
// Flash attention with analytic 7x7 neighborhood mask.
// qkv: (B*1024, 768) [q|k|v], ctx: (B*1024, 256)
// grid: (1024/16, NH=8, B=8), block 32 (one wave / 16-query tile).
// A 16-query tile lies in one feature-map row qy, so the mask restricts keys
// to the 32-aligned band rows [qy-3, qy+3]: <=224 keys instead of 1024.
// V tiles staged global->LDS with async b128 copies (ASYNCcnt).
// ---------------------------------------------------------------------------
__global__ void attn_wmma_kernel(const float* __restrict__ qkv,
                                 float* __restrict__ ctx) {
  const int lane = threadIdx.x;
  const int g = lane >> 4, nn = lane & 15;
  const int q0 = blockIdx.x * 16;
  const int h = blockIdx.y;
  const int b = blockIdx.z;
  const float scale = 0.17677669529663687f;    // 1/sqrt(32)

  const float* __restrict__ qrow =
      qkv + ((size_t)(b * 1024 + q0 + nn)) * 768 + h * 32;
  const v16h aq = load_frag_rowK(qrow, 0, 32, g, true);

  v8f acc0 = {}, acc1 = {};
  float mrun[8], lrun[8];
#pragma unroll
  for (int f = 0; f < 8; ++f) { mrun[f] = -3.0e38f; lrun[f] = 0.f; }

  __shared__ _Float16 plds[16][32];
  __shared__ float vtile[32][36];   // 144B rows: 16B-aligned, bank-clean

  const int qy = q0 >> 5;                       // tile's feature-map row
  const int klo = (qy > 3 ? qy - 3 : 0) * 32;   // 32-aligned key band
  const int khi = (qy + 4 < 32 ? qy + 4 : 32) * 32;

  for (int kt = klo; kt < khi; kt += 32) {
    // stage V tile (32 keys x 32 dh, f32) into LDS via async copies
#pragma unroll
    for (int i = 0; i < 8; ++i) {
      const int chunk = i * 32 + lane;          // 256 x 16B chunks
      const int key = chunk >> 3;
      const int c4 = (chunk & 7) * 4;
      const float* gv =
          qkv + ((size_t)(b * 1024 + kt + key)) * 768 + 512 + h * 32 + c4;
      unsigned lv = (unsigned)(uintptr_t)&vtile[key][c4];
      asm volatile("global_load_async_to_lds_b128 %0, %1, off"
                   :: "v"(lv), "v"(gv) : "memory");
    }

    v8f s[2];
#pragma unroll
    for (int t = 0; t < 2; ++t) {
      const int k0 = kt + t * 16;
      const float* __restrict__ krow =
          qkv + ((size_t)(b * 1024 + k0 + nn)) * 768 + 256 + h * 32;
      v16h bk = load_frag_rowK(krow, 0, 32, g, true);
      v8f z = {};
      s[t] = __builtin_amdgcn_wmma_f32_16x16x32_f16(false, aq, false, bk,
                                                    (short)0, z, false, false);
      const int key = k0 + nn;
      const int ky = key >> 5, kx = key & 31;
#pragma unroll
      for (int f = 0; f < 8; ++f) {
        const int q = q0 + f + 8 * g;
        const int qx = q & 31;
        bool allowed = (abs(qy - ky) <= 3) && (abs(qx - kx) <= 3);
        s[t][f] = s[t][f] * scale + (allowed ? 0.f : -1e9f);
      }
    }

    // online softmax: rows striped across the two 16-lane groups
#pragma unroll
    for (int f = 0; f < 8; ++f) {
      float v = fmaxf(s[0][f], s[1][f]);
      for (int off = 1; off < 16; off <<= 1)
        v = fmaxf(v, __shfl_xor(v, off, 16));
      const float mnew = fmaxf(mrun[f], v);
      const float alpha = expf(mrun[f] - mnew);
      const float p0 = expf(s[0][f] - mnew);
      const float p1 = expf(s[1][f] - mnew);
      float rs = p0 + p1;
      for (int off = 1; off < 16; off <<= 1)
        rs += __shfl_xor(rs, off, 16);
      lrun[f] = lrun[f] * alpha + rs;
      mrun[f] = mnew;
      acc0[f] *= alpha;
      acc1[f] *= alpha;
      plds[f + 8 * g][nn]      = (_Float16)p0;
      plds[f + 8 * g][nn + 16] = (_Float16)p1;
    }
    // single wave: DS store->load in order (dscnt waits inserted by compiler);
    // async V copies completed before LDS reads below:
    asm volatile("s_wait_asynccnt 0x0" ::: "memory");

    v16h ap, bv0, bv1;
#pragma unroll
    for (int e = 0; e < 16; ++e) {
      int k = g * 8 + e + ((e < 8) ? 0 : 8);
      ap[e]  = plds[nn][k];
      bv0[e] = (_Float16)vtile[k][nn];
      bv1[e] = (_Float16)vtile[k][nn + 16];
    }
    acc0 = __builtin_amdgcn_wmma_f32_16x16x32_f16(false, ap, false, bv0,
                                                  (short)0, acc0, true, false);
    acc1 = __builtin_amdgcn_wmma_f32_16x16x32_f16(false, ap, false, bv1,
                                                  (short)0, acc1, false, false);
  }

#pragma unroll
  for (int f = 0; f < 8; ++f) {
    const int q = q0 + f + 8 * g;
    const float inv = 1.f / lrun[f];
    float* __restrict__ o = ctx + ((size_t)(b * 1024 + q)) * 256 + h * 32;
    o[nn]      = acc0[f] * inv;
    o[nn + 16] = acc1[f] * inv;
  }
}

// ---------------------------------------------------------------------------
// LayerNorm over C=256: one row per block, 256 threads.
// ---------------------------------------------------------------------------
__global__ void ln_kernel(const float* __restrict__ in,
                          const float* __restrict__ gw,
                          const float* __restrict__ gb,
                          float* __restrict__ out) {
  const int row = blockIdx.x, t = threadIdx.x;
  const float v = in[(size_t)row * 256 + t];
  __shared__ float s1[256], s2[256];
  s1[t] = v; s2[t] = v * v;
  __syncthreads();
  for (int s = 128; s > 0; s >>= 1) {
    if (t < s) { s1[t] += s1[t + s]; s2[t] += s2[t + s]; }
    __syncthreads();
  }
  const float mu = s1[0] * (1.f / 256.f);
  const float var = s2[0] * (1.f / 256.f) - mu * mu;
  const float inv = rsqrtf(var + 1e-5f);
  out[(size_t)row * 256 + t] = (v - mu) * inv * gw[t] + gb[t];
}

// ---------------------------------------------------------------------------
// Sine positional embedding: pos[n, c], n in [0,1024), c in [0,256)
// ---------------------------------------------------------------------------
__global__ void pos_kernel(float* __restrict__ pos) {
  const int idx = blockIdx.x * blockDim.x + threadIdx.x;
  if (idx >= 1024 * 256) return;
  const int n = idx >> 8, c = idx & 255;
  const float y = (float)(n >> 5) + 1.f;
  const float x = (float)(n & 31) + 1.f;
  const float coord = (c < 128) ? y : x;
  const int cc = (c < 128) ? c : c - 128;
  const float denom = powf(10000.f, (float)(2 * (cc >> 1)) * (1.f / 128.f));
  const float ang = coord / denom;
  pos[idx] = (cc & 1) ? cosf(ang) : sinf(ang);
}

// qk_in = src + pos (broadcast over batch)
__global__ void add_pos_kernel(const float* __restrict__ src,
                               const float* __restrict__ pos,
                               float* __restrict__ out) {
  const int idx = blockIdx.x * blockDim.x + threadIdx.x;
  if (idx >= 8192 * 256) return;
  const int n = (idx >> 8) & 1023;
  const int c = idx & 255;
  out[idx] = src[idx] + pos[n * 256 + c];
}

// (B,256,1024) -> (B,1024,256)
__global__ void chw_to_nc(const float* __restrict__ x, float* __restrict__ src) {
  const int idx = blockIdx.x * blockDim.x + threadIdx.x;
  if (idx >= 8 * 256 * 1024) return;
  const int b = idx >> 18, rem = idx & 262143;
  const int c = rem >> 10, n = rem & 1023;
  src[((size_t)b * 1024 + n) * 256 + c] = x[idx];
}

// (B,1024,256) -> (B,256,32,32)
__global__ void nc_to_chw(const float* __restrict__ src, float* __restrict__ out) {
  const int idx = blockIdx.x * blockDim.x + threadIdx.x;
  if (idx >= 8 * 256 * 1024) return;
  const int b = idx >> 18, rem = idx & 262143;
  const int c = rem >> 10, n = rem & 1023;
  out[idx] = src[((size_t)b * 1024 + n) * 256 + c];
}

// ---------------------------------------------------------------------------
extern "C" void kernel_launch(void* const* d_in, const int* in_sizes, int n_in,
                              void* d_out, int out_size, void* d_ws, size_t ws_size,
                              hipStream_t stream) {
  (void)in_sizes; (void)n_in; (void)out_size; (void)ws_size;
  const float* hint = (const float*)d_in[0];
  const float* cw[8];
  for (int i = 0; i < 8; ++i) cw[i] = (const float*)d_in[1 + i];
  const float* ipw = (const float*)d_in[9];
  const float* ipb = (const float*)d_in[10];
  const float* oww = (const float*)d_in[11];
  const float* owb = (const float*)d_in[12];
  const float* l1w = (const float*)d_in[13];
  const float* l1b = (const float*)d_in[14];
  const float* l2w = (const float*)d_in[15];
  const float* l2b = (const float*)d_in[16];
  const float* g1 = (const float*)d_in[17];
  const float* b1 = (const float*)d_in[18];
  const float* g2 = (const float*)d_in[19];
  const float* b2 = (const float*)d_in[20];

  float* ws = (float*)d_ws;
  const size_t SZBUF = 8192ull * 1025ull;      // >= largest conv buffer
  float* buf0 = ws;                             // conv ping A / FFN hidden
  float* buf1 = buf0 + SZBUF;                   // conv ping B / QKV
  float* src  = buf1 + SZBUF;                   // (8192, 256) token stream
  float* posb = src + 8192ull * 256;            // (1024, 256)
  float* ctxb = posb + 1024ull * 256;           // (8192, 256)
  float* tmpb = ctxb + 8192ull * 256;           // (8192, 256) qk_in / pre-LN

  // ---- conv stack ----
  struct CC { int co, ci, s; };
  const CC cfg[8] = {{16,3,1},{16,16,1},{32,16,2},{32,32,1},
                     {96,32,2},{96,96,1},{256,96,2},{256,256,1}};
  const float* cin = hint;
  int Hi = 256;
  float* bufs[2] = {buf0, buf1};
  for (int i = 0; i < 8; ++i) {
    const int Ho = (Hi - 1) / cfg[i].s + 1;
    const int Np = Ho * Ho;
    float* cout = bufs[i & 1];
    dim3 grid(Np / 32, cfg[i].co / 16, 8);
    conv_wmma_kernel<<<grid, 32, 0, stream>>>(cin, cw[i], cout,
                                              cfg[i].ci, Hi, Hi,
                                              cfg[i].co, Ho, Ho,
                                              cfg[i].s, (i < 7) ? 1 : 0);
    cin = cout;
    Hi = Ho;
  }
  // conv7 output lives in buf1: (8,256,32,32)
  chw_to_nc<<<(8 * 256 * 1024 + 255) / 256, 256, 0, stream>>>(buf1, src);
  pos_kernel<<<1024, 256, 0, stream>>>(posb);

  const int M = 8192, Cc = 256, DFF = 1025;
  float* qkv = buf1;    // (8192, 768)  -- conv buffers are free now
  float* ff  = buf0;    // (8192, 1025)

  for (int l = 0; l < 8; ++l) {
    const float* Wip = ipw + (size_t)l * 768 * 256;
    const float* Bip = ipb + (size_t)l * 768;
    const float* Wo  = oww + (size_t)l * 256 * 256;
    const float* Bo  = owb + (size_t)l * 256;
    const float* W1  = l1w + (size_t)l * DFF * 256;
    const float* B1  = l1b + (size_t)l * DFF;
    const float* W2  = l2w + (size_t)l * 256 * DFF;
    const float* B2  = l2b + (size_t)l * 256;
    const float* G1l = g1 + (size_t)l * 256; const float* B1l = b1 + (size_t)l * 256;
    const float* G2l = g2 + (size_t)l * 256; const float* B2l = b2 + (size_t)l * 256;

    // qk_in = src + pos; then Q,K projections from qk_in, V from src
    add_pos_kernel<<<(8192 * 256 + 255) / 256, 256, 0, stream>>>(src, posb, tmpb);
    gemm_wmma_kernel<<<dim3((512 + 63) / 64, M / 64), 256, 0, stream>>>(
        tmpb, Cc, Wip, Bip, nullptr, qkv, 768, M, 512, Cc, 0);
    gemm_wmma_kernel<<<dim3((256 + 63) / 64, M / 64), 256, 0, stream>>>(
        src, Cc, Wip + 512 * 256, Bip + 512, nullptr,
        qkv + 512, 768, M, 256, Cc, 0);

    // masked flash attention (banded)
    attn_wmma_kernel<<<dim3(64, 8, 8), 32, 0, stream>>>(qkv, ctxb);

    // out projection + residual, then LN1
    gemm_wmma_kernel<<<dim3((256 + 63) / 64, M / 64), 256, 0, stream>>>(
        ctxb, Cc, Wo, Bo, src, tmpb, Cc, M, Cc, Cc, 0);
    ln_kernel<<<M, 256, 0, stream>>>(tmpb, G1l, B1l, src);

    // FFN: relu GEMM -> GEMM + residual, then LN2
    gemm_wmma_kernel<<<dim3((DFF + 63) / 64, M / 64), 256, 0, stream>>>(
        src, Cc, W1, B1, nullptr, ff, DFF, M, DFF, Cc, 1);
    gemm_wmma_kernel<<<dim3((256 + 63) / 64, M / 64), 256, 0, stream>>>(
        ff, DFF, W2, B2, src, tmpb, Cc, M, Cc, DFF, 0);
    ln_kernel<<<M, 256, 0, stream>>>(tmpb, G2l, B2l, src);
  }

  nc_to_chw<<<(8 * 256 * 1024 + 255) / 256, 256, 0, stream>>>(src, (float*)d_out);
}
